// LSTM_14259291423388
// MI455X (gfx1250) — compile-verified, matrix operand order checked by reading, not run on hardware
//
#include <hip/hip_runtime.h>
#include <hip/hip_bf16.h>

typedef __bf16 bf16_t;
typedef __attribute__((ext_vector_type(16))) __bf16 v16bf;
typedef __attribute__((ext_vector_type(8)))  __bf16 v8bf;
typedef __attribute__((ext_vector_type(4)))  __bf16 v4bf;
typedef __attribute__((ext_vector_type(8)))  float  v8f;

#define LSTM_B 64
#define LSTM_S 512
#define LSTM_D 1024
#define LSTM_H 1024
#define LSTM_K 2048   // D + H (both layers: in + H = 2048)
#define LSTM_G 4096   // 4*H

union FragU { v16bf v; v8bf h[2]; };

__device__ __forceinline__ v16bf load_frag(const bf16_t* __restrict__ base,
                                           int off0, int off1) {
  FragU u;
  u.h[0] = *(const v8bf*)(base + off0);   // 16B aligned chunks (K multiples of 8)
  u.h[1] = *(const v8bf*)(base + off1);
  return u.v;
}

__device__ __forceinline__ float fsigmoid(float x) {
  return 1.0f / (1.0f + __expf(-x));
}
__device__ __forceinline__ float ftanh(float x) {
  // tanh(x) = 1 - 2/(exp(2x)+1); saturates correctly for |x| large
  return 1.0f - 2.0f / (__expf(2.0f * x) + 1.0f);
}

// Fused GEMM(WMMA bf16) + LSTM cell.
// gates[m,n] = sum_k combined[m,k] * W[n,k] + bias[n]; n split i/f/g/o by 1024.
// One wave: 16 rows (mtile) x 16 cols (jtile) x 4 gates -> full cell update.
__global__ __launch_bounds__(256) void lstm_cell_wmma(
    const bf16_t* __restrict__ A0,     // rows [64 x 1024], k in [0,1024)
    long a0_stride,                    // row stride of A0 in elements
    const bf16_t* __restrict__ A1,     // h_prev of this layer [64 x 1024]
    const bf16_t* __restrict__ W,      // [4096 x 2048] bf16 row-major
    const float*  __restrict__ bias,   // [4096]
    float*  __restrict__ c_state,      // [64 x 1024] in-place (element-owned)
    float*  __restrict__ h_state,      // [64 x 1024] fp32 (for h_fin)
    bf16_t* __restrict__ h_out_bf,     // [64 x 1024] write-parity bf16 state
    float*  __restrict__ out_h,        // nullptr, or out + t*H
    long out_stride)                   // S*H
{
  const int lane  = threadIdx.x & 31;
  const int wave  = threadIdx.x >> 5;
  const int gw    = blockIdx.x * (blockDim.x >> 5) + wave;  // 0..255
  const int mtile = gw & 3;     // 4 tiles of M=64
  const int jtile = gw >> 2;    // 64 tiles of H=1024
  const int hsel  = lane >> 4;  // ISA A/B half-wave K select
  const int l16   = lane & 15;
  const int kh    = hsel * 8;

  const int jglob = jtile * 16 + l16;         // B/C/D column for this lane
  const int arow  = mtile * 16 + l16;         // A row for this lane

  const bf16_t* a0row = A0 + (long)arow * a0_stride;
  const bf16_t* a1row = A1 + (long)arow * LSTM_H;

  const bf16_t* wrow[4];
  v8f acc[4];
#pragma unroll
  for (int g = 0; g < 4; ++g) {
    wrow[g] = W + (long)(g * LSTM_H + jglob) * LSTM_K;
    float bv = bias[g * LSTM_H + jglob];      // same col for all 8 rows
    acc[g] = (v8f){bv, bv, bv, bv, bv, bv, bv, bv};
  }

  // K in [0,1024): input source (x_t for layer0, h_layer0(t) for layer1)
#pragma unroll 2
  for (int k0 = 0; k0 < LSTM_D; k0 += 32) {
    v16bf af = load_frag(a0row, k0 + kh, k0 + 16 + kh);
#pragma unroll
    for (int g = 0; g < 4; ++g) {
      v16bf bm = load_frag(wrow[g], k0 + kh, k0 + 16 + kh);
      acc[g] = __builtin_amdgcn_wmma_f32_16x16x32_bf16(
          false, af, false, bm, (short)0, acc[g], false, false);
    }
  }
  // K in [1024,2048): this layer's h_{t-1}
#pragma unroll 2
  for (int k0 = 0; k0 < LSTM_H; k0 += 32) {
    v16bf af = load_frag(a1row, k0 + kh, k0 + 16 + kh);
#pragma unroll
    for (int g = 0; g < 4; ++g) {
      v16bf bm = load_frag(wrow[g], LSTM_D + k0 + kh, LSTM_D + k0 + 16 + kh);
      acc[g] = __builtin_amdgcn_wmma_f32_16x16x32_bf16(
          false, af, false, bm, (short)0, acc[g], false, false);
    }
  }

  // Fused cell epilogue. C/D layout: lane<16 -> M=r, lane>=16 -> M=r+8.
#pragma unroll
  for (int r = 0; r < 8; ++r) {
    const int m  = mtile * 16 + r + hsel * 8;
    const long idx = (long)m * LSTM_H + jglob;
    const float ig = fsigmoid(acc[0][r]);
    const float fg = fsigmoid(acc[1][r]);
    const float gg = ftanh(acc[2][r]);
    const float og = fsigmoid(acc[3][r]);
    const float cn = fg * c_state[idx] + ig * gg;
    const float hn = og * ftanh(cn);
    c_state[idx]  = cn;
    h_state[idx]  = hn;
    h_out_bf[idx] = (bf16_t)hn;
    if (out_h) out_h[(long)m * out_stride + jglob] = hn;
  }
}

__global__ void cvt_f32_to_bf16(const float* __restrict__ in,
                                bf16_t* __restrict__ out, long n4) {
  long i = (long)blockIdx.x * blockDim.x + threadIdx.x;
  if (i < n4) {
    const float4 v = *(const float4*)(in + i * 4);
    v4bf o;
    o[0] = (bf16_t)v.x; o[1] = (bf16_t)v.y;
    o[2] = (bf16_t)v.z; o[3] = (bf16_t)v.w;
    *(v4bf*)(out + i * 4) = o;
  }
}

__global__ void zero_u32(unsigned* __restrict__ p, long n) {
  long i = (long)blockIdx.x * blockDim.x + threadIdx.x;
  if (i < n) p[i] = 0u;
}

__global__ void copy_f32(const float* __restrict__ src,
                         float* __restrict__ dst, long n) {
  long i = (long)blockIdx.x * blockDim.x + threadIdx.x;
  if (i < n) dst[i] = src[i];
}

extern "C" void kernel_launch(void* const* d_in, const int* in_sizes, int n_in,
                              void* d_out, int out_size, void* d_ws, size_t ws_size,
                              hipStream_t stream) {
  (void)in_sizes; (void)n_in; (void)out_size; (void)ws_size;
  const float* x  = (const float*)d_in[0];   // [64,512,1024]
  const float* W0 = (const float*)d_in[1];   // [4096,2048]
  const float* b0 = (const float*)d_in[2];   // [4096]
  const float* W1 = (const float*)d_in[3];   // [4096,2048]
  const float* b1 = (const float*)d_in[4];   // [4096]
  float* out = (float*)d_out;                // [64,512,1024] ++ h_fin[2,64,1024] ++ c_fin[2,64,1024]

  const long BH   = (long)LSTM_B * LSTM_H;           // 65536
  const long WN   = (long)LSTM_G * LSTM_K;           // 8388608 per layer
  const long XN   = (long)LSTM_B * LSTM_S * LSTM_D;  // 33554432

  // Workspace layout
  bf16_t* Wbf  = (bf16_t*)d_ws;            // [2][4096][2048] bf16
  bf16_t* xbf  = Wbf + 2 * WN;             // [64][512][1024] bf16
  float*  c_st = (float*)(xbf + XN);       // [2][64][1024] f32
  float*  h_st = c_st + 2 * BH;            // [2][64][1024] f32
  bf16_t* hbf  = (bf16_t*)(h_st + 2 * BH); // [layer][parity][64][1024] bf16

  // 1) Precision conversion (weights stay L2-resident as bf16 thereafter)
  {
    const int T = 256;
    cvt_f32_to_bf16<<<(unsigned)((WN / 4 + T - 1) / T), T, 0, stream>>>(W0, Wbf, WN / 4);
    cvt_f32_to_bf16<<<(unsigned)((WN / 4 + T - 1) / T), T, 0, stream>>>(W1, Wbf + WN, WN / 4);
    cvt_f32_to_bf16<<<(unsigned)((XN / 4 + T - 1) / T), T, 0, stream>>>(x, xbf, XN / 4);
  }
  // 2) Zero all recurrent state (c, h_f32, both bf16 parities) -- deterministic per call
  {
    const long nwords = (2 * BH + 2 * BH) + (4 * BH * (long)sizeof(bf16_t)) / 4;
    zero_u32<<<(unsigned)((nwords + 255) / 256), 256, 0, stream>>>((unsigned*)c_st, nwords);
  }

  // 3) Sequential scan: 512 steps x 2 layers, each a fused WMMA GEMM + cell
  for (int t = 0; t < LSTM_S; ++t) {
    const int pr = t & 1;        // read parity
    const int pw = 1 - pr;       // write parity
    // layer 0: A0 = x[:, t, :], A1 = h0_{t-1}
    lstm_cell_wmma<<<32, 256, 0, stream>>>(
        xbf + (long)t * LSTM_D, (long)LSTM_S * LSTM_D,
        hbf + (0 * 2 + pr) * BH,
        Wbf, b0,
        c_st, h_st,
        hbf + (0 * 2 + pw) * BH,
        nullptr, 0);
    // layer 1: A0 = h0_t (just written), A1 = h1_{t-1}; emits output row t
    lstm_cell_wmma<<<32, 256, 0, stream>>>(
        hbf + (0 * 2 + pw) * BH, (long)LSTM_H,
        hbf + (1 * 2 + pr) * BH,
        Wbf + WN, b1,
        c_st + BH, h_st + BH,
        hbf + (1 * 2 + pw) * BH,
        out + (long)t * LSTM_H, (long)LSTM_S * LSTM_H);
  }

  // 4) Final states: h_fin [2,64,1024] then c_fin [2,64,1024]
  {
    float* tail = out + (long)LSTM_B * LSTM_S * LSTM_H;
    copy_f32<<<(unsigned)((2 * BH + 255) / 256), 256, 0, stream>>>(h_st, tail, 2 * BH);
    copy_f32<<<(unsigned)((2 * BH + 255) / 256), 256, 0, stream>>>(c_st, tail + 2 * BH, 2 * BH);
  }
}
